// MultiViewAttention_69475390980202
// MI455X (gfx1250) — matrix operand order repaired and checked by reading, not compiled, not run on hardware
//
#include <hip/hip_runtime.h>
#include <hip/hip_bf16.h>

typedef _Float16 v16h __attribute__((ext_vector_type(16)));
typedef _Float16 v8h  __attribute__((ext_vector_type(8)));
typedef float    v8f  __attribute__((ext_vector_type(8)));

// ---------------------------------------------------------------------------
// WMMA helpers (CDNA5 gfx1250, wave32). D = A(16x32,f16) * B(32x16,f16) + C(f32)
// ---------------------------------------------------------------------------
__device__ __forceinline__ v8f wmma_f16(v16h a, v16h b, v8f c) {
    return __builtin_amdgcn_wmma_f32_16x16x32_f16(
        /*neg_a=*/false, a, /*neg_b=*/false, b,
        /*c_mod=*/(short)0, c, /*reuse_a=*/false, /*reuse_b=*/false);
}

// A fragment: matrix X [M,K] row-major (ld = row stride in elements).
// lane m = lane&15 ; halves at k-offsets koff..koff+7 and 16+koff..16+koff+7,
// koff = (lane>>4)*8  (per ISA 16-bit A 16x32 layout).
__device__ __forceinline__ v16h load_a_frag(const _Float16* base, int ld,
                                            int m_base, int k_base, int lane) {
    int m    = m_base + (lane & 15);
    int koff = (lane >> 4) * 8;
    const _Float16* p = base + (size_t)m * ld + k_base + koff;
    v8h lo = *(const v8h*)p;
    v8h hh = *(const v8h*)(p + 16);
    v16h r;
#pragma unroll
    for (int i = 0; i < 8; i++) { r[i] = lo[i]; r[i + 8] = hh[i]; }
    return r;
}

// B fragment for C = A * W^T with W stored [N,K] row-major (torch-Linear form).
// lane n = lane&15 loads row n at k-offsets koff..koff+7 and 16+koff..16+koff+7, koff=(lane>>4)*8.
__device__ __forceinline__ v16h load_b_frag(const _Float16* base, int ld,
                                            int n_base, int k_base, int lane) {
    int n  = n_base + (lane & 15);
    int kk = k_base + (lane >> 4) * 8;
    const _Float16* p = base + (size_t)n * ld + kk;
    v8h lo = *(const v8h*)p;
    v8h hh = *(const v8h*)(p + 16);
    v16h r;
#pragma unroll
    for (int i = 0; i < 8; i++) { r[i] = lo[i]; r[i + 8] = hh[i]; }
    return r;
}

__device__ __forceinline__ float half_wave_max(float v) {
#pragma unroll
    for (int off = 1; off < 16; off <<= 1) v = fmaxf(v, __shfl_xor(v, off, 32));
    return v;
}
__device__ __forceinline__ float half_wave_sum(float v) {
#pragma unroll
    for (int off = 1; off < 16; off <<= 1) v += __shfl_xor(v, off, 32);
    return v;
}

// ---------------------------------------------------------------------------
// fp32 -> f16 conversion
// ---------------------------------------------------------------------------
__global__ void cvt_f32_to_f16(const float* __restrict__ in,
                               _Float16* __restrict__ out, int n) {
    int i = blockIdx.x * blockDim.x + threadIdx.x;
    if (i < n) out[i] = (_Float16)in[i];
}

// ---------------------------------------------------------------------------
// QKV GEMM: qkv[8192, 2304] = X[8192,768] @ Wqkv[2304,768]^T + bqkv
// Epilogue scatters into q[B,h,S,64], k[B,h,S,64], vT[B,h,64,S]  (all f16).
// Block: 256 thr = 8 waves; wave tile 16(M)x64(N); block tile 128x64.
// ---------------------------------------------------------------------------
__global__ __launch_bounds__(256) void qkv_gemm(
    const _Float16* __restrict__ X, const _Float16* __restrict__ W,
    const float* __restrict__ bias,
    _Float16* __restrict__ qbuf, _Float16* __restrict__ kbuf,
    _Float16* __restrict__ vT) {
    const int lane = threadIdx.x & 31;
    const int wave = threadIdx.x >> 5;
    const int m0 = blockIdx.x * 128 + wave * 16;
    const int n0 = blockIdx.y * 64;

    v8f acc[4] = {};
    for (int k = 0; k < 768; k += 32) {
        v16h a = load_a_frag(X, 768, m0, k, lane);
#pragma unroll
        for (int t = 0; t < 4; t++) {
            v16h b = load_b_frag(W, 768, n0 + 16 * t, k, lane);
            acc[t] = wmma_f16(a, b, acc[t]);
        }
    }

    const int hi = lane >> 4, nlo = lane & 15;
#pragma unroll
    for (int t = 0; t < 4; t++) {
        int gn   = n0 + 16 * t + nlo;
        float bv = bias[gn];
        int head = gn / 192, rem = gn % 192;
        int part = rem / 64, dd = rem % 64;   // part uniform within a 16-tile
#pragma unroll
        for (int r = 0; r < 8; r++) {
            int m = m0 + 8 * hi + r;
            int b_ = m >> 10, s = m & 1023;
            _Float16 val = (_Float16)(acc[t][r] + bv);
            size_t bh = (size_t)(b_ * 12 + head);
            if (part == 0)      qbuf[(bh * 1024 + s) * 64 + dd] = val;
            else if (part == 1) kbuf[(bh * 1024 + s) * 64 + dd] = val;
            else                vT[(bh * 64 + dd) * 1024 + s]   = val;
        }
    }
}

// ---------------------------------------------------------------------------
// Flash attention per (b,h): scores = q k^T / 8, key-mask -> -1e9, online
// softmax, o = p v. Block: 128 thr = 4 waves; wave owns 16 q-rows; grid.y
// tiles S/64. p tile converted D-layout -> A-layout through a 1KB LDS tile.
// ---------------------------------------------------------------------------
__global__ __launch_bounds__(128) void flash_attn(
    const _Float16* __restrict__ q, const _Float16* __restrict__ k,
    const _Float16* __restrict__ vT, const int* __restrict__ mask,
    _Float16* __restrict__ aout) {
    __shared__ __align__(16) _Float16 pbuf[4][16][32];

    const int lane = threadIdx.x & 31;
    const int wave = threadIdx.x >> 5;
    const int bh = blockIdx.x;            // 0..95
    const int b  = bh / 12, head = bh % 12;
    const int q0 = blockIdx.y * 64 + wave * 16;
    const int hi = lane >> 4, nlo = lane & 15;

    const _Float16* qm = q  + (size_t)bh * 1024 * 64;
    const _Float16* km = k  + (size_t)bh * 1024 * 64;
    const _Float16* vm = vT + (size_t)bh * 64 * 1024;
    const int* mk = mask + b * 1024;

    v16h qa0 = load_a_frag(qm, 64, q0, 0,  lane);
    v16h qa1 = load_a_frag(qm, 64, q0, 32, lane);

    v8f o[4] = {};
    float mrun[8], lrun[8];
#pragma unroll
    for (int r = 0; r < 8; r++) { mrun[r] = -1e30f; lrun[r] = 0.0f; }

    for (int j0 = 0; j0 < 1024; j0 += 32) {
        v8f s[2];
#pragma unroll
        for (int t = 0; t < 2; t++) {
            v16h kb0 = load_b_frag(km, 64, j0 + 16 * t, 0,  lane);
            v16h kb1 = load_b_frag(km, 64, j0 + 16 * t, 32, lane);
            v8f z = {};
            z    = wmma_f16(qa0, kb0, z);
            s[t] = wmma_f16(qa1, kb1, z);
            bool msk = mk[j0 + 16 * t + nlo] != 0;   // one key-column per lane
#pragma unroll
            for (int r = 0; r < 8; r++)
                s[t][r] = msk ? -1e9f : s[t][r] * 0.125f;
        }
        // online softmax over this 32-wide key slab
#pragma unroll
        for (int r = 0; r < 8; r++) {
            float mx   = half_wave_max(fmaxf(s[0][r], s[1][r]));
            float newm = fmaxf(mrun[r], mx);
            float p0 = __expf(s[0][r] - newm);
            float p1 = __expf(s[1][r] - newm);
            s[0][r] = p0; s[1][r] = p1;
            float sum   = half_wave_sum(p0 + p1);
            float scale = __expf(mrun[r] - newm);
            lrun[r] = lrun[r] * scale + sum;
            mrun[r] = newm;
#pragma unroll
            for (int t2 = 0; t2 < 4; t2++) o[t2][r] *= scale;
        }
        // D-layout -> A-layout via LDS (wave-private tile)
#pragma unroll
        for (int t = 0; t < 2; t++)
#pragma unroll
            for (int r = 0; r < 8; r++)
                pbuf[wave][8 * hi + r][16 * t + nlo] = (_Float16)s[t][r];
        __builtin_amdgcn_wave_barrier();
        asm volatile("s_wait_dscnt 0" ::: "memory");
        v16h pa;
        {
            const _Float16* pp = &pbuf[wave][nlo][0];
            v8h lo = *(const v8h*)(pp + hi * 8);
            v8h hh = *(const v8h*)(pp + 16 + hi * 8);
#pragma unroll
            for (int i = 0; i < 8; i++) { pa[i] = lo[i]; pa[i + 8] = hh[i]; }
        }
        __builtin_amdgcn_wave_barrier();
#pragma unroll
        for (int t = 0; t < 4; t++) {
            v16h vb = load_b_frag(vm, 1024, 16 * t, j0, lane); // rows of v^T
            o[t] = wmma_f16(pa, vb, o[t]);
        }
    }
    // finalize: divide by l, scatter to attn_out[B,S,768] as f16
#pragma unroll
    for (int t = 0; t < 4; t++) {
        int dd = 16 * t + nlo;
#pragma unroll
        for (int r = 0; r < 8; r++) {
            int srow = q0 + 8 * hi + r;
            aout[((size_t)(b * 1024 + srow)) * 768 + head * 64 + dd] =
                (_Float16)(o[t][r] / lrun[r]);
        }
    }
}

// ---------------------------------------------------------------------------
// Output GEMM: out[8192,768] = A[8192,768] @ Wo[768,768]^T + bo  (fp32 out)
// ---------------------------------------------------------------------------
__global__ __launch_bounds__(256) void out_gemm(
    const _Float16* __restrict__ A, const _Float16* __restrict__ W,
    const float* __restrict__ bias, float* __restrict__ out) {
    const int lane = threadIdx.x & 31;
    const int wave = threadIdx.x >> 5;
    const int m0 = blockIdx.x * 128 + wave * 16;
    const int n0 = blockIdx.y * 64;

    v8f acc[4] = {};
    for (int k = 0; k < 768; k += 32) {
        v16h a = load_a_frag(A, 768, m0, k, lane);
#pragma unroll
        for (int t = 0; t < 4; t++) {
            v16h b = load_b_frag(W, 768, n0 + 16 * t, k, lane);
            acc[t] = wmma_f16(a, b, acc[t]);
        }
    }
    const int hi = lane >> 4, nlo = lane & 15;
#pragma unroll
    for (int t = 0; t < 4; t++) {
        int gn = n0 + 16 * t + nlo;
        float bv = bias[gn];
#pragma unroll
        for (int r = 0; r < 8; r++) {
            int m = m0 + 8 * hi + r;
            out[(size_t)m * 768 + gn] = acc[t][r] + bv;
        }
    }
}

// ---------------------------------------------------------------------------
// Launcher. B=8, V=4, F=256, H=768 -> S=1024, heads=12, d=64.
// ---------------------------------------------------------------------------
extern "C" void kernel_launch(void* const* d_in, const int* in_sizes, int n_in,
                              void* d_out, int out_size, void* d_ws, size_t ws_size,
                              hipStream_t stream) {
    const float* x    = (const float*)d_in[0];   // [8,4,256,768]
    const int*   mask = (const int*)d_in[1];     // [8,4,256] (nonzero = masked)
    const float* Wqkv = (const float*)d_in[2];   // [2304,768]
    const float* bqkv = (const float*)d_in[3];   // [2304]
    const float* Wo   = (const float*)d_in[4];   // [768,768]
    const float* bo   = (const float*)d_in[5];   // [768]
    float* out = (float*)d_out;

    const int NX    = 8192 * 768;    // 6,291,456
    const int NWQKV = 2304 * 768;    // 1,769,472
    const int NWO   = 768 * 768;     //   589,824
    const int NQKV  = 8 * 12 * 1024 * 64; // 6,291,456 per tensor

    _Float16* xh    = (_Float16*)d_ws;
    _Float16* wqkvh = xh    + NX;
    _Float16* woh   = wqkvh + NWQKV;
    _Float16* qb    = woh   + NWO;
    _Float16* kb    = qb    + NQKV;
    _Float16* vT    = kb    + NQKV;
    _Float16* ao    = vT    + NQKV;  // total ~67.6 MB of workspace

    // 1) convert inputs to f16
    cvt_f32_to_f16<<<(NX + 255) / 256, 256, 0, stream>>>(x, xh, NX);
    cvt_f32_to_f16<<<(NWQKV + 255) / 256, 256, 0, stream>>>(Wqkv, wqkvh, NWQKV);
    cvt_f32_to_f16<<<(NWO + 255) / 256, 256, 0, stream>>>(Wo, woh, NWO);

    // 2) fused QKV projection (M=8192, N=2304, K=768)
    qkv_gemm<<<dim3(8192 / 128, 2304 / 64), 256, 0, stream>>>(
        xh, wqkvh, bqkv, qb, kb, vT);

    // 3) flash attention: grid (B*h, S/64), 4 waves x 16 q-rows
    flash_attn<<<dim3(96, 16), 128, 0, stream>>>(qb, kb, vT, mask, ao);

    // 4) output projection (M=8192, N=768, K=768) -> fp32 d_out
    out_gemm<<<dim3(8192 / 128, 768 / 64), 256, 0, stream>>>(ao, woh, bo, out);
}
